// IPF_22299470201448
// MI455X (gfx1250) — matrix operand order repaired and checked by reading, not compiled, not run on hardware
//
#include <hip/hip_runtime.h>

typedef _Float16 h16;
typedef __attribute__((ext_vector_type(16))) _Float16 v16h;
typedef __attribute__((ext_vector_type(8)))  float    v8f;

#define PI_F       3.14159265358979323846f
#define FOV_UP_F   0.034906585039886586f
#define FOV_DOWN_F (-0.43284165446944687f)

union AF { unsigned u[8]; v16h v; };

__device__ __forceinline__ v8f v8f_zero() {
  v8f z = {0.f,0.f,0.f,0.f,0.f,0.f,0.f,0.f};
  return z;
}

__device__ __forceinline__ v8f wmma_f16(v16h a, v16h b, v8f c) {
  return __builtin_amdgcn_wmma_f32_16x16x32_f16(false, a, false, b, (short)0, c, false, false);
}

// 16-bit A-matrix 16x32 layout (contiguous k): lanes 0-15 hold M=lane; VGPR0..7 -> K 0..15;
// lanes 16-31 add +16 to K.
__device__ __forceinline__ int a_frag_k(int lane, int h) {
  int vg = h >> 1, lo = h & 1;
  int k = 2*vg + lo;
  return k + ((lane & 16) ? 16 : 0);
}

// rowPtr = &A[row*lda + kc*32], must be 4B aligned; loads merge into 2x ds_load_b128.
__device__ __forceinline__ v16h load_a_frag(const h16* rowPtr, int lane) {
  AF af;
  #pragma unroll
  for (int vg = 0; vg < 8; ++vg) {
    int k = a_frag_k(lane, vg*2);
    af.u[vg] = *(const unsigned*)(rowPtr + k);
  }
  return af.v;
}

__device__ __forceinline__ void calc_ray(float c0, float c1, float* r) {
  float pitch = FOV_UP_F + (c0 + 1.f) * 0.5f * (FOV_DOWN_F - FOV_UP_F);
  float yaw = c1 * PI_F;
  float cp = __cosf(pitch);
  r[0] = cp * __cosf(yaw);
  r[1] = cp * __sinf(yaw);
  r[2] = __sinf(pitch);
}

// embed(v): [v0,v1,v2, then per freq j: sin(v*2^j) x3, cos(v*2^j) x3] -> 63 halves
__device__ __forceinline__ void embed_store(float x0, float x1, float x2, h16* dst) {
  dst[0] = (h16)x0; dst[1] = (h16)x1; dst[2] = (h16)x2;
  float f = 1.f;
  #pragma unroll
  for (int j = 0; j < 10; ++j) {
    float a0 = x0*f, a1 = x1*f, a2 = x2*f;
    dst[3 + j*6 + 0] = (h16)__sinf(a0);
    dst[3 + j*6 + 1] = (h16)__sinf(a1);
    dst[3 + j*6 + 2] = (h16)__sinf(a2);
    dst[3 + j*6 + 3] = (h16)__cosf(a0);
    dst[3 + j*6 + 4] = (h16)__cosf(a1);
    dst[3 + j*6 + 5] = (h16)__cosf(a2);
    f *= 2.f;
  }
}

// ---------------------------------------------------------------------------
// Weight packing: WMMA-B lane layout.
// dst[((kc*ntT + nt)*32 + lane)*16 + h] = src[k= kc*32 + h + 16*(lane>=16)][n= nt*16 + (lane&15)]
// ---------------------------------------------------------------------------
__global__ void pack_b_dense(const float* __restrict__ src, h16* __restrict__ dst,
                             int K, int N, int ldn, int kcT, int ntT) {
  int idx = blockIdx.x * blockDim.x + threadIdx.x;
  int total = kcT * ntT * 512;
  if (idx >= total) return;
  int h    = idx & 15;
  int lane = (idx >> 4) & 31;
  int rest = idx >> 9;
  int nt   = rest % ntT;
  int kc   = rest / ntT;
  int k = kc*32 + h + ((lane & 16) ? 16 : 0);
  int n = nt*16 + (lane & 15);
  float v = (k < K && n < N) ? src[k*ldn + n] : 0.f;
  dst[idx] = (h16)v;
}

// Conv weights (O=64, I=64, 3, 3): B(k = tap*64 + ci, n = o) = w[o][ci][tap]
__global__ void pack_b_conv(const float* __restrict__ src, h16* __restrict__ dst) {
  int idx = blockIdx.x * blockDim.x + threadIdx.x;
  const int total = 18 * 4 * 512;   // K=576, N=64
  if (idx >= total) return;
  int h    = idx & 15;
  int lane = (idx >> 4) & 31;
  int rest = idx >> 9;
  int nt   = rest & 3;
  int kc   = rest >> 2;
  int k = kc*32 + h + ((lane & 16) ? 16 : 0);
  int n = nt*16 + (lane & 15);
  int tap = k / 64, ci = k % 64;
  dst[idx] = (h16)src[n*576 + ci*9 + tap];
}

// ---------------------------------------------------------------------------
// Head conv: 1 -> 64 ch, 3x3 SAME. VALU (tiny). Output f16 HWC.
// ---------------------------------------------------------------------------
__global__ void __launch_bounds__(256) conv_head(const float* __restrict__ inp,
                                                 const float* __restrict__ w,
                                                 const float* __restrict__ b,
                                                 h16* __restrict__ X) {
  __shared__ float sw[576];
  __shared__ float sb[64];
  for (int i = threadIdx.x; i < 576; i += 256) sw[i] = w[i];
  if (threadIdx.x < 64) sb[threadIdx.x] = b[threadIdx.x];
  __syncthreads();
  int p = blockIdx.x * 256 + threadIdx.x;
  int y = p >> 10, x = p & 1023;
  float vin[9];
  #pragma unroll
  for (int t = 0; t < 9; ++t) {
    int yy = y + t/3 - 1, xx = x + t%3 - 1;
    vin[t] = (yy >= 0 && yy < 64 && xx >= 0 && xx < 1024) ? inp[yy*1024 + xx] : 0.f;
  }
  for (int c = 0; c < 64; ++c) {
    float acc = sb[c];
    #pragma unroll
    for (int t = 0; t < 9; ++t) acc += vin[t] * sw[c*9 + t];
    X[(size_t)p*64 + c] = (h16)acc;
  }
}

// ---------------------------------------------------------------------------
// 64->64 ch 3x3 conv, implicit GEMM via WMMA. feature layout f16 HWC.
// Halo tile staged to LDS with CDNA5 async global->LDS DMA (ASYNCcnt).
// WG = 256 thr / 8 waves; tile = 1 row x 64 px; wave = (mt = w&3, nt pair).
// ---------------------------------------------------------------------------
__global__ void __launch_bounds__(256) conv64(const h16* __restrict__ in,
                                              const h16* __restrict__ wpk,
                                              const float* __restrict__ bias,
                                              const h16* __restrict__ res,
                                              h16* __restrict__ out, int relu) {
  __shared__ __align__(16) h16 sIn[3*68*64];
  int tid = threadIdx.x;
  int y  = blockIdx.x >> 4;
  int x0 = (blockIdx.x & 15) * 64;

  for (int s = tid; s < 3*66; s += 256) {
    int ry = s / 66, xo = s % 66;
    int gy = y - 1 + ry, gx = x0 - 1 + xo;
    h16* dsth = &sIn[(ry*68 + xo)*64];
    if (gy >= 0 && gy < 64 && gx >= 0 && gx < 1024) {
      const h16* srch = in + ((size_t)gy*1024 + gx)*64;
      // 128 bytes per pixel slot: 8x async b128 direct to LDS (no VGPR bounce).
      unsigned laddr =
          (unsigned)(unsigned long long)(__attribute__((address_space(3))) h16*)dsth;
      unsigned long long gaddr = (unsigned long long)srch;
      #pragma unroll
      for (int i = 0; i < 8; ++i) {
        asm volatile("global_load_async_to_lds_b128 %0, %1, off"
                     :: "v"(laddr + i*16u), "v"(gaddr + (unsigned long long)(i*16))
                     : "memory");
      }
    } else {
      uint4 z = make_uint4(0u,0u,0u,0u);
      uint4* dst = (uint4*)dsth;
      #pragma unroll
      for (int i = 0; i < 8; ++i) dst[i] = z;
    }
  }
  asm volatile("s_wait_asynccnt 0" ::: "memory");
  __syncthreads();

  int w = tid >> 5, lane = tid & 31;
  int mt  = w & 3;
  int ntb = (w >> 2) * 2;
  int col = lane & 15;
  int rhi = (lane & 16) ? 8 : 0;
  int px  = mt*16 + col;          // A row = pixel in tile

  v8f acc0 = v8f_zero(), acc1 = v8f_zero();
  for (int kc = 0; kc < 18; ++kc) {
    int tap = kc >> 1;
    int ry = tap / 3, dx = tap % 3;
    int cb = (kc & 1) * 32;
    v16h a = load_a_frag(&sIn[(ry*68 + px + dx)*64 + cb], lane);
    const h16* bp = wpk + ((kc*4 + ntb)*32 + lane)*16;
    v16h b0 = *(const v16h*)(bp);
    v16h b1 = *(const v16h*)(bp + 512);  // next nt: +32*16 halves
    if (kc + 1 < 18) __builtin_prefetch(wpk + (((kc+1)*4 + ntb)*32 + lane)*16, 0, 0);
    acc0 = wmma_f16(a, b0, acc0);
    acc1 = wmma_f16(a, b1, acc1);
  }

  #pragma unroll
  for (int j = 0; j < 2; ++j) {
    v8f a = j ? acc1 : acc0;
    int c = (ntb + j)*16 + col;
    float bb = bias[c];
    #pragma unroll
    for (int r = 0; r < 8; ++r) {
      int m = mt*16 + r + rhi;
      size_t p = ((size_t)y*1024 + x0 + m)*64 + c;
      float v = a[r] + bb;
      if (relu) v = fmaxf(v, 0.f);
      if (res)  v += (float)res[p];
      __builtin_nontemporal_store((h16)v, &out[p]);
    }
  }
}

// ---------------------------------------------------------------------------
// Fused gather + embedding + 4-layer MLP + lin head. 64 points per WG.
// ---------------------------------------------------------------------------
struct MParams {
  const float* inp; const float* coord; const h16* feat;
  const h16* w0; const float* b0;
  const h16* w1; const float* b1;
  const h16* w2; const float* b2;
  const h16* w3; const float* b3;
  const h16* lw; const float* lb;
  h16* tok; float* p4;
};

__device__ __forceinline__ void mlp_layer(const h16* A, h16* O, const h16* Bpk,
                                          const float* bias, bool relu, int tid) {
  int w = tid >> 5, lane = tid & 31;
  int mt = w & 3, half = w >> 2;
  int row = lane & 15;
  int rhi = (lane & 16) ? 8 : 0;
  v8f accs[8];
  #pragma unroll
  for (int j = 0; j < 8; ++j) accs[j] = v8f_zero();
  for (int kc = 0; kc < 8; ++kc) {
    v16h a = load_a_frag(&A[(mt*16 + row)*256 + kc*32], lane);
    #pragma unroll
    for (int j = 0; j < 8; ++j) {
      int nt = half*8 + j;
      v16h b = *(const v16h*)(Bpk + ((kc*16 + nt)*32 + lane)*16);
      accs[j] = wmma_f16(a, b, accs[j]);
    }
  }
  #pragma unroll
  for (int j = 0; j < 8; ++j) {
    int c = (half*8 + j)*16 + row;
    float bb = bias[c];
    #pragma unroll
    for (int r = 0; r < 8; ++r) {
      float v = accs[j][r] + bb;
      if (relu) v = fmaxf(v, 0.f);
      O[(mt*16 + r + rhi)*256 + c] = (h16)v;
    }
  }
}

__device__ __forceinline__ void mlp_layer_final(const h16* A, float* O, const h16* Bpk,
                                                const float* bias, int tid) {
  int w = tid >> 5, lane = tid & 31;
  int mt = w & 3, half = w >> 2;
  int row = lane & 15;
  int rhi = (lane & 16) ? 8 : 0;
  int ntBase = half * 3;   // half0: nt 0,1,2 ; half1: nt 3,4
  v8f accs[3];
  #pragma unroll
  for (int j = 0; j < 3; ++j) accs[j] = v8f_zero();
  for (int kc = 0; kc < 8; ++kc) {
    v16h a = load_a_frag(&A[(mt*16 + row)*256 + kc*32], lane);
    #pragma unroll
    for (int j = 0; j < 3; ++j) {
      int nt = ntBase + j;
      if (nt < 5) {   // wave-uniform
        v16h b = *(const v16h*)(Bpk + ((kc*5 + nt)*32 + lane)*16);
        accs[j] = wmma_f16(a, b, accs[j]);
      }
    }
  }
  #pragma unroll
  for (int j = 0; j < 3; ++j) {
    int nt = ntBase + j;
    if (nt < 5) {
      int c = nt*16 + row;
      float bb = (c < 65) ? bias[c] : 0.f;
      #pragma unroll
      for (int r = 0; r < 8; ++r)
        O[(mt*16 + r + rhi)*80 + c] = accs[j][r] + bb;
    }
  }
}

__device__ __forceinline__ void lin_layer(const h16* A, const float* outL, const h16* qf,
                                          const h16* Bpk, const float* lb,
                                          h16* tokG, int base, int tid) {
  int w = tid >> 5, lane = tid & 31;
  int mt = w & 3, half = w >> 2;
  int row = lane & 15;
  int rhi = (lane & 16) ? 8 : 0;
  v8f accs[2];
  accs[0] = v8f_zero(); accs[1] = v8f_zero();
  for (int kc = 0; kc < 2; ++kc) {
    v16h a = load_a_frag(&A[(mt*16 + row)*64 + kc*32], lane);
    #pragma unroll
    for (int j = 0; j < 2; ++j) {
      int nt = half*2 + j;
      v16h b = *(const v16h*)(Bpk + ((kc*4 + nt)*32 + lane)*16);
      accs[j] = wmma_f16(a, b, accs[j]);
    }
  }
  #pragma unroll
  for (int j = 0; j < 2; ++j) {
    int c = (half*2 + j)*16 + row;
    float bb = lb[c];
    #pragma unroll
    for (int r = 0; r < 8; ++r) {
      int m = mt*16 + r + rhi;
      float v = (float)qf[m*64 + c] + outL[m*80 + 1 + c] + accs[j][r] + bb;
      __builtin_nontemporal_store((h16)v, &tokG[((size_t)(base + m))*64 + c]);
    }
  }
}

__global__ void __launch_bounds__(256) mlp_fused(MParams mp) {
  __shared__ __align__(16) h16 act0[64*256];
  __shared__ __align__(16) h16 act1[64*256];
  __shared__ __align__(16) float outL[64*80];
  __shared__ __align__(16) h16 sQf[64*64];
  __shared__ float sRay[64][3];
  __shared__ float sPd[64];
  __shared__ float sRR[64];
  __shared__ int   sPix[64];

  int tid = threadIdx.x;
  int base = blockIdx.x * 64;

  if (tid < 64) {
    int P = base + tid;
    int q = P >> 2, o = P & 3;
    float c0 = mp.coord[q*2 + 0], c1 = mp.coord[q*2 + 1];
    float ray[3]; calc_ray(c0, c1, ray);
    embed_store(ray[0], ray[1], ray[2], &act0[tid*256]);       // dir_emb [0..62]

    float ox = (o & 2) ? 1.f : -1.f;
    float oy = (o & 1) ? 1.f : -1.f;
    float cc0 = c0 + ox*(1.f/64.f)   + 1e-6f;
    float cc1 = c1 + oy*(1.f/1024.f) + 1e-6f;
    cc0 = fminf(fmaxf(cc0, -1.f + 1e-6f), 1.f - 1e-6f);
    cc1 = fminf(fmaxf(cc1, -1.f + 1e-6f), 1.f - 1e-6f);
    int iy = (int)floorf((cc0 + 1.f) * 32.f);  iy = iy < 0 ? 0 : (iy > 63 ? 63 : iy);
    int ix = (int)floorf((cc1 + 1.f) * 512.f); ix = ix < 0 ? 0 : (ix > 1023 ? 1023 : ix);
    int pix = iy*1024 + ix;
    sPix[tid] = pix;

    float r = mp.inp[pix];
    float qc0 = -1.f + ((float)iy + 0.5f) * (2.f/64.f);
    float qc1 = -1.f + ((float)ix + 0.5f) * (2.f/1024.f);
    float qray[3]; calc_ray(qc0, qc1, qray);
    float rr = (r + 1.f) * 0.5f;
    float qp0 = rr*qray[0], qp1 = rr*qray[1], qp2 = rr*qray[2];
    float pd = qp0*ray[0] + qp1*ray[1] + qp2*ray[2];
    float pj0 = pd*ray[0], pj1 = pd*ray[1], pj2 = pd*ray[2];
    embed_store(-(qp0-pj0), -(qp1-pj1), -(qp2-pj2), &act0[tid*256 + 63]);   // embed(-rej)
    embed_store(pj0, pj1, pj2, &act0[tid*256 + 126]);                        // embed(proj)
    sRay[tid][0] = ray[0]; sRay[tid][1] = ray[1]; sRay[tid][2] = ray[2];
    sPd[tid] = pd; sRR[tid] = rr;
    act0[tid*256 + 253] = (h16)0.f;
    act0[tid*256 + 254] = (h16)0.f;
    act0[tid*256 + 255] = (h16)0.f;
  }
  __syncthreads();
  {
    int p = tid >> 2, part = tid & 3;
    int pix = sPix[p];
    const h16* src = mp.feat + (size_t)pix*64 + part*16;
    #pragma unroll
    for (int i = 0; i < 16; ++i) {
      h16 v = src[i];
      act0[p*256 + 189 + part*16 + i] = v;   // q_feat [189..252]
      sQf[p*64 + part*16 + i] = v;
    }
  }
  __syncthreads();

  mlp_layer(act0, act1, mp.w0, mp.b0, true, tid);  __syncthreads();
  mlp_layer(act1, act0, mp.w1, mp.b1, true, tid);  __syncthreads();
  mlp_layer(act0, act1, mp.w2, mp.b2, true, tid);  __syncthreads();
  mlp_layer_final(act1, outL, mp.w3, mp.b3, tid);  __syncthreads();

  if (tid < 64) {
    float pred = sPd[tid] + outL[tid*80 + 0];
    if (sRR[tid] + 1e-6f > 1.0f) pred = 1.0f;
    float pp0 = pred*sRay[tid][0], pp1 = pred*sRay[tid][1], pp2 = pred*sRay[tid][2];
    mp.p4[base + tid] = pred*2.f - 1.f;
    embed_store(pp0, pp1, pp2, &act0[tid*64]);   // reuse act0 as [64][64]
    act0[tid*64 + 63] = (h16)0.f;
  }
  __syncthreads();
  lin_layer(act0, outL, sQf, mp.lw, mp.lb, mp.tok, base, tid);
}

// ---------------------------------------------------------------------------
// Transformer over (B*Q, 4, 64) token groups. 1 wave per 4 groups (M=16).
// ---------------------------------------------------------------------------
struct TParams {
  const h16* tok; const float* p4; float* out;
  const h16* qkvpk[2]; const h16* aopk[2]; const h16* f1pk[2]; const h16* f2pk[2];
  const float* ln1g[2]; const float* ln1b[2]; const float* aob[2];
  const float* ln2g[2]; const float* ln2b[2]; const float* fb1[2]; const float* fb2[2];
  const float* hg; const float* hb; const float* hw; const float* hbias;
};

// Hoisted A-fragments (K=64 => 2 fragments), streamed B tiles.
__device__ __forceinline__ void load_a_pair(const h16* A, int lane, v16h& a0, v16h& a1) {
  int row = lane & 15;
  a0 = load_a_frag(&A[row*64], lane);
  a1 = load_a_frag(&A[row*64 + 32], lane);
}

__device__ __forceinline__ v8f gemm16_pre(v16h a0, v16h a1, const h16* Bpk,
                                          int ntT, int nt, int lane) {
  v8f acc = v8f_zero();
  v16h b0 = *(const v16h*)(Bpk + ((0*ntT + nt)*32 + lane)*16);
  v16h b1 = *(const v16h*)(Bpk + ((1*ntT + nt)*32 + lane)*16);
  acc = wmma_f16(a0, b0, acc);
  acc = wmma_f16(a1, b1, acc);
  return acc;
}

__device__ __forceinline__ void store_frag_h16(h16* dst, int ld, int nt, v8f acc, int lane) {
  int col = nt*16 + (lane & 15);
  int rhi = (lane & 16) ? 8 : 0;
  #pragma unroll
  for (int r = 0; r < 8; ++r) dst[(r + rhi)*ld + col] = (h16)acc[r];
}

__device__ __forceinline__ void store_frag_gelu(h16* dst, int ld, int nt, v8f acc,
                                                const float* bias, int lane) {
  int col = nt*16 + (lane & 15);
  int rhi = (lane & 16) ? 8 : 0;
  float bb = bias[col];
  #pragma unroll
  for (int r = 0; r < 8; ++r) {
    float x = acc[r] + bb;
    float t = tanhf(0.7978845608f * (x + 0.044715f*x*x*x));
    dst[(r + rhi)*ld + col] = (h16)(0.5f*x*(1.f + t));
  }
}

__device__ __forceinline__ void add_frag_f32(float* dst, int ld, int nt, v8f acc,
                                             const float* bias, int lane) {
  int col = nt*16 + (lane & 15);
  int rhi = (lane & 16) ? 8 : 0;
  float bb = bias[col];
  #pragma unroll
  for (int r = 0; r < 8; ++r) dst[(r + rhi)*ld + col] += acc[r] + bb;
}

__device__ __forceinline__ void ln_into(h16* dst, const float* tok, const float* g,
                                        const float* b, float* mu, float* rs, int lane) {
  if (lane < 16) {
    float s = 0.f, ss = 0.f;
    for (int c = 0; c < 64; ++c) { float v = tok[lane*64 + c]; s += v; ss += v*v; }
    float m = s * (1.f/64.f);
    float var = ss * (1.f/64.f) - m*m;
    mu[lane] = m; rs[lane] = rsqrtf(var + 1e-5f);
  }
  for (int i = lane; i < 1024; i += 32) {
    int row = i >> 6, c = i & 63;
    dst[i] = (h16)((tok[i] - mu[row]) * rs[row] * g[c] + b[c]);
  }
}

__device__ __forceinline__ void attn4(const h16* qkv, h16* obuf, int lane) {
  int g = lane >> 3, hd = lane & 7;
  const float scale = 0.35355339059327373f;   // 8^-0.5
  float sc[4][4];
  #pragma unroll
  for (int i = 0; i < 4; ++i) {
    #pragma unroll
    for (int j = 0; j < 4; ++j) {
      float s = 0.f;
      #pragma unroll
      for (int d = 0; d < 8; ++d)
        s += (float)qkv[(g*4+i)*192 + hd*8 + d] * (float)qkv[(g*4+j)*192 + 64 + hd*8 + d];
      sc[i][j] = s * scale;
    }
  }
  #pragma unroll
  for (int i = 0; i < 4; ++i) {
    float m = fmaxf(fmaxf(sc[i][0], sc[i][1]), fmaxf(sc[i][2], sc[i][3]));
    float e0 = __expf(sc[i][0]-m), e1 = __expf(sc[i][1]-m);
    float e2 = __expf(sc[i][2]-m), e3 = __expf(sc[i][3]-m);
    float inv = 1.f / (e0 + e1 + e2 + e3);
    #pragma unroll
    for (int d = 0; d < 8; ++d) {
      float o = e0*(float)qkv[(g*4+0)*192 + 128 + hd*8 + d]
              + e1*(float)qkv[(g*4+1)*192 + 128 + hd*8 + d]
              + e2*(float)qkv[(g*4+2)*192 + 128 + hd*8 + d]
              + e3*(float)qkv[(g*4+3)*192 + 128 + hd*8 + d];
      obuf[(g*4+i)*64 + hd*8 + d] = (h16)(o * inv);
    }
  }
}

__global__ void __launch_bounds__(256) transformer_kernel(TParams tp) {
  __shared__ __align__(16) float sTok[8][1024];
  __shared__ __align__(16) h16   sBuf[8][1024];
  __shared__ __align__(16) h16   sQKV[8][3072];
  __shared__ float sMu[8][16], sRs[8][16], sLg[8][16];
  __shared__ float sP[2][7][64];   // ln1g,ln1b,aob,ln2g,ln2b,fb1,fb2
  __shared__ float sHg[64], sHb[64], sHw[64];
  __shared__ float sHbias;

  int tid = threadIdx.x;
  if (tid < 64) {
    #pragma unroll
    for (int l = 0; l < 2; ++l) {
      sP[l][0][tid] = tp.ln1g[l][tid]; sP[l][1][tid] = tp.ln1b[l][tid];
      sP[l][2][tid] = tp.aob[l][tid];
      sP[l][3][tid] = tp.ln2g[l][tid]; sP[l][4][tid] = tp.ln2b[l][tid];
      sP[l][5][tid] = tp.fb1[l][tid];  sP[l][6][tid] = tp.fb2[l][tid];
    }
    sHg[tid] = tp.hg[tid]; sHb[tid] = tp.hb[tid]; sHw[tid] = tp.hw[tid];
    if (tid == 0) sHbias = tp.hbias[0];
  }
  __syncthreads();

  int w = tid >> 5, lane = tid & 31;
  int q0 = blockIdx.x*32 + w*4;
  size_t tbase = (size_t)q0 * 256;     // 4 tokens * 64 dims per group
  float* tok = sTok[w];
  h16* buf = sBuf[w];
  h16* qkv = sQKV[w];

  for (int i = lane; i < 1024; i += 32) tok[i] = (float)tp.tok[tbase + i];

  for (int l = 0; l < 2; ++l) {
    ln_into(buf, tok, sP[l][0], sP[l][1], sMu[w], sRs[w], lane);
    {
      v16h a0, a1; load_a_pair(buf, lane, a0, a1);
      for (int nt = 0; nt < 12; ++nt) {
        v8f acc = gemm16_pre(a0, a1, tp.qkvpk[l], 12, nt, lane);
        store_frag_h16(qkv, 192, nt, acc, lane);
      }
    }
    attn4(qkv, buf, lane);
    {
      v16h a0, a1; load_a_pair(buf, lane, a0, a1);
      for (int nt = 0; nt < 4; ++nt) {
        v8f acc = gemm16_pre(a0, a1, tp.aopk[l], 4, nt, lane);
        add_frag_f32(tok, 64, nt, acc, sP[l][2], lane);
      }
    }
    ln_into(buf, tok, sP[l][3], sP[l][4], sMu[w], sRs[w], lane);
    {
      v16h a0, a1; load_a_pair(buf, lane, a0, a1);
      for (int nt = 0; nt < 4; ++nt) {
        v8f acc = gemm16_pre(a0, a1, tp.f1pk[l], 4, nt, lane);
        store_frag_gelu(qkv, 64, nt, acc, sP[l][5], lane);   // reuse qkv as FF buffer
      }
    }
    {
      v16h a0, a1; load_a_pair(qkv, lane, a0, a1);
      for (int nt = 0; nt < 4; ++nt) {
        v8f acc = gemm16_pre(a0, a1, tp.f2pk[l], 4, nt, lane);
        add_frag_f32(tok, 64, nt, acc, sP[l][6], lane);
      }
    }
  }

  if (lane < 16) {
    float s = 0.f, ss = 0.f;
    for (int c = 0; c < 64; ++c) { float v = tok[lane*64 + c]; s += v; ss += v*v; }
    float m = s * (1.f/64.f);
    float var = ss * (1.f/64.f) - m*m;
    float rs = rsqrtf(var + 1e-5f);
    float lg = sHbias;
    for (int c = 0; c < 64; ++c)
      lg += ((tok[lane*64 + c] - m) * rs * sHg[c] + sHb[c]) * sHw[c];
    sLg[w][lane] = lg;
  }
  if (lane < 4) {
    int g = lane;
    float m = fmaxf(fmaxf(sLg[w][g*4+0], sLg[w][g*4+1]), fmaxf(sLg[w][g*4+2], sLg[w][g*4+3]));
    float e0 = __expf(sLg[w][g*4+0]-m), e1 = __expf(sLg[w][g*4+1]-m);
    float e2 = __expf(sLg[w][g*4+2]-m), e3 = __expf(sLg[w][g*4+3]-m);
    float inv = 1.f / (e0 + e1 + e2 + e3);
    const float* p4 = tp.p4 + (size_t)(q0 + g)*4;
    tp.out[q0 + g] = (p4[0]*e0 + p4[1]*e1 + p4[2]*e2 + p4[3]*e3) * inv;
  }
}

// ---------------------------------------------------------------------------
// Host-side orchestration
// ---------------------------------------------------------------------------
extern "C" void kernel_launch(void* const* d_in, const int* in_sizes, int n_in,
                              void* d_out, int out_size, void* d_ws, size_t ws_size,
                              hipStream_t stream) {
  (void)in_sizes; (void)n_in; (void)out_size; (void)ws_size;

  const float* inp   = (const float*)d_in[0];
  const float* coord = (const float*)d_in[1];
  const float* enc_head_w = (const float*)d_in[2];
  const float* enc_head_b = (const float*)d_in[3];
  const float* enc_w1 = (const float*)d_in[4];
  const float* enc_b1 = (const float*)d_in[5];
  const float* enc_w2 = (const float*)d_in[6];
  const float* enc_b2 = (const float*)d_in[7];
  const float* enc_tail_w = (const float*)d_in[8];
  const float* enc_tail_b = (const float*)d_in[9];
  const float* mlp_w[4] = {(const float*)d_in[10], (const float*)d_in[12],
                           (const float*)d_in[14], (const float*)d_in[16]};
  const float* mlp_b[4] = {(const float*)d_in[11], (const float*)d_in[13],
                           (const float*)d_in[15], (const float*)d_in[17]};
  const float* lin_w = (const float*)d_in[18];
  const float* lin_b = (const float*)d_in[19];
  const float* head_ln_g = (const float*)d_in[20];
  const float* head_ln_b = (const float*)d_in[21];
  const float* head_w = (const float*)d_in[22];
  const float* head_b = (const float*)d_in[23];
  const float *ln1g[2], *ln1b[2], *qkvw[2], *aow[2], *aob[2];
  const float *ln2g[2], *ln2b[2], *fw1[2], *fb1[2], *fw2[2], *fb2[2];
  for (int l = 0; l < 2; ++l) {
    int o = 24 + l*11;
    ln1g[l] = (const float*)d_in[o+0]; ln1b[l] = (const float*)d_in[o+1];
    qkvw[l] = (const float*)d_in[o+2]; aow[l]  = (const float*)d_in[o+3];
    aob[l]  = (const float*)d_in[o+4];
    ln2g[l] = (const float*)d_in[o+5]; ln2b[l] = (const float*)d_in[o+6];
    fw1[l]  = (const float*)d_in[o+7]; fb1[l]  = (const float*)d_in[o+8];
    fw2[l]  = (const float*)d_in[o+9]; fb2[l]  = (const float*)d_in[o+10];
  }

  // ---- workspace carve (bytes, 256-aligned) ----
  char* ws = (char*)d_ws;
  size_t off = 0;
  auto alloc = [&](size_t bytes) -> char* {
    char* p = ws + off;
    off = (off + bytes + 255) & ~(size_t)255;
    return p;
  };
  const size_t FEAT_B = (size_t)64*1024*64*sizeof(h16);   // 8 MB
  h16* X  = (h16*)alloc(FEAT_B);
  h16* H0 = (h16*)alloc(FEAT_B);
  h16* H1 = (h16*)alloc(FEAT_B);
  h16* T  = (h16*)alloc(FEAT_B);
  const int PKC = 18*4*512;                               // 36864 halves / conv
  h16* PW1 = (h16*)alloc((size_t)16*PKC*sizeof(h16));
  h16* PW2 = (h16*)alloc((size_t)16*PKC*sizeof(h16));
  h16* PT  = (h16*)alloc((size_t)PKC*sizeof(h16));
  h16* PM[4];
  PM[0] = (h16*)alloc((size_t)8*16*512*sizeof(h16));
  PM[1] = (h16*)alloc((size_t)8*16*512*sizeof(h16));
  PM[2] = (h16*)alloc((size_t)8*16*512*sizeof(h16));
  PM[3] = (h16*)alloc((size_t)8*5*512*sizeof(h16));
  h16* PLIN = (h16*)alloc((size_t)2*4*512*sizeof(h16));
  h16 *PQKV[2], *PAO[2], *PF1[2], *PF2[2];
  for (int l = 0; l < 2; ++l) {
    PQKV[l] = (h16*)alloc((size_t)2*12*512*sizeof(h16));
    PAO[l]  = (h16*)alloc((size_t)2*4*512*sizeof(h16));
    PF1[l]  = (h16*)alloc((size_t)2*4*512*sizeof(h16));
    PF2[l]  = (h16*)alloc((size_t)2*4*512*sizeof(h16));
  }
  h16*  TOK = (h16*)alloc((size_t)262144*64*sizeof(h16)); // 32 MB
  float* P4 = (float*)alloc((size_t)262144*sizeof(float));

  // ---- weight packing ----
  for (int i = 0; i < 16; ++i) {
    pack_b_conv<<<144, 256, 0, stream>>>(enc_w1 + (size_t)i*36864, PW1 + (size_t)i*PKC);
    pack_b_conv<<<144, 256, 0, stream>>>(enc_w2 + (size_t)i*36864, PW2 + (size_t)i*PKC);
  }
  pack_b_conv<<<144, 256, 0, stream>>>(enc_tail_w, PT);

  auto packDense = [&](const float* src, h16* dst, int K, int N, int ldn, int kcT, int ntT) {
    int total = kcT*ntT*512;
    pack_b_dense<<<(total + 255)/256, 256, 0, stream>>>(src, dst, K, N, ldn, kcT, ntT);
  };
  packDense(mlp_w[0], PM[0], 253, 256, 256, 8, 16);
  packDense(mlp_w[1], PM[1], 256, 256, 256, 8, 16);
  packDense(mlp_w[2], PM[2], 256, 256, 256, 8, 16);
  packDense(mlp_w[3], PM[3], 256, 65, 65, 8, 5);
  packDense(lin_w, PLIN, 63, 64, 64, 2, 4);
  for (int l = 0; l < 2; ++l) {
    packDense(qkvw[l], PQKV[l], 64, 192, 192, 2, 12);
    packDense(aow[l],  PAO[l],  64, 64, 64, 2, 4);
    packDense(fw1[l],  PF1[l],  64, 64, 64, 2, 4);
    packDense(fw2[l],  PF2[l],  64, 64, 64, 2, 4);
  }

  // ---- encoder ----
  conv_head<<<256, 256, 0, stream>>>(inp, enc_head_w, enc_head_b, X);

  const h16* cur = X;
  h16* bufs[2] = {H0, H1};
  int pb = 0;
  for (int i = 0; i < 16; ++i) {
    conv64<<<1024, 256, 0, stream>>>(cur, PW1 + (size_t)i*PKC, enc_b1 + i*64,
                                     (const h16*)nullptr, T, 1);
    h16* nxt = bufs[pb]; pb ^= 1;
    conv64<<<1024, 256, 0, stream>>>(T, PW2 + (size_t)i*PKC, enc_b2 + i*64,
                                     cur, nxt, 0);
    cur = nxt;
  }
  h16* FEAT = bufs[pb];     // free buffer
  conv64<<<1024, 256, 0, stream>>>(cur, PT, enc_tail_b, X, FEAT, 0);

  // ---- fused gather + MLP ----
  MParams mp;
  mp.inp = inp; mp.coord = coord; mp.feat = FEAT;
  mp.w0 = PM[0]; mp.b0 = mlp_b[0];
  mp.w1 = PM[1]; mp.b1 = mlp_b[1];
  mp.w2 = PM[2]; mp.b2 = mlp_b[2];
  mp.w3 = PM[3]; mp.b3 = mlp_b[3];
  mp.lw = PLIN;  mp.lb = lin_b;
  mp.tok = TOK;  mp.p4 = P4;
  mlp_fused<<<4096, 256, 0, stream>>>(mp);

  // ---- transformer + head ----
  TParams tp;
  tp.tok = TOK; tp.p4 = P4; tp.out = (float*)d_out;
  for (int l = 0; l < 2; ++l) {
    tp.qkvpk[l] = PQKV[l]; tp.aopk[l] = PAO[l]; tp.f1pk[l] = PF1[l]; tp.f2pk[l] = PF2[l];
    tp.ln1g[l] = ln1g[l]; tp.ln1b[l] = ln1b[l]; tp.aob[l] = aob[l];
    tp.ln2g[l] = ln2g[l]; tp.ln2b[l] = ln2b[l]; tp.fb1[l] = fb1[l]; tp.fb2[l] = fb2[l];
  }
  tp.hg = head_ln_g; tp.hb = head_ln_b; tp.hw = head_w; tp.hbias = head_b;
  transformer_kernel<<<2048, 256, 0, stream>>>(tp);
}
